// RPNClassificationAndRegressionLoss_16810501997265
// MI455X (gfx1250) — compile-verified, matrix operand order checked by reading, not run on hardware
//
#include <hip/hip_runtime.h>
#include <hip/hip_bf16.h>
#include <math.h>

#define M_PROPS   200000
#define N_GT      128
#define POS_THR   0.4f
#define NEG_THR   0.1f
#define REG_FACT  10.0f

#define SALT_POS  0x13572468u
#define SALT_NEG  0x2468ACEFu
#define SALT_BKT  0x0BADF00Du
#define SALT_BEST 0x9E3779B9u

typedef __attribute__((ext_vector_type(16))) _Float16     v16h;
typedef __attribute__((ext_vector_type(8)))  float        v8f;
typedef __attribute__((ext_vector_type(4))) unsigned int  u32x4;
typedef __attribute__((ext_vector_type(8)))  int          i32x8;
typedef __attribute__((ext_vector_type(4)))  int          i32x4;

__device__ __forceinline__ unsigned hash32(unsigned x, unsigned salt) {
    x ^= salt;
    x *= 0x9E3779B9u; x ^= x >> 16;
    x *= 0x85EBCA6Bu; x ^= x >> 13;
    x *= 0xC2B2AE35u; x ^= x >> 16;
    return x;
}

__device__ __forceinline__ unsigned long long packKey(unsigned hi, unsigned lo) {
    return ((unsigned long long)hi << 32) | (unsigned long long)lo;
}

__device__ __forceinline__ float smooth_l1(float d) {
    float ad = fabsf(d);
    return (ad < 1.0f) ? 0.5f * d * d : ad - 0.5f;
}

// Matrix-unit reduction: sum of 512 f16 values (16 per lane, 32 lanes) via
// D(16x16,f32) = A(16x32,f16) x ones(32x16,f16). Every D row is the row-sum of A
// replicated across 16 columns, so total(A) = sum(D)/16.
__device__ __forceinline__ float wmma_sum512(v16h a, v16h ones) {
    v8f c = {};
    c = __builtin_amdgcn_wmma_f32_16x16x32_f16(
            /*neg_a=*/false, a, /*neg_b=*/false, ones,
            /*c_mod=*/(short)0, c, /*reuse_a=*/false, /*reuse_b=*/false);
    float s = 0.0f;
    #pragma unroll
    for (int i = 0; i < 8; ++i) s += c[i];
    #pragma unroll
    for (int off = 16; off > 0; off >>= 1) s += __shfl_xor(s, off, 32);
    return s * (1.0f / 16.0f);
}

// ---------------------------------------------------------------------------
// Kernel 1: zero the 384 u64 reduction slots (128 row-best, 128 pos, 128 neg)
// ---------------------------------------------------------------------------
__global__ void init_ws(unsigned long long* ws) {
    int i = threadIdx.x;
    if (i < 3 * N_GT) ws[i] = 0ull;
}

// ---------------------------------------------------------------------------
// Kernel 2: IoU + deterministic sampling. One thread per proposal. GT tile is
// staged into LDS via the Tensor Data Mover (2D descriptor: 128 rows x 4 f32).
// ---------------------------------------------------------------------------
__global__ void __launch_bounds__(256)
iou_sample(const float4* __restrict__ proposals,
           const float4* __restrict__ gt,
           unsigned long long* __restrict__ ws) {
    __shared__ float4 gtS[N_GT];                       // 2 KB, TDM destination
    __shared__ unsigned long long rowS[N_GT];          // per-row best (iou,col)
    __shared__ unsigned long long posS[N_GT];          // positive sample buckets
    __shared__ unsigned long long negS[N_GT];          // negative sample buckets

    const int tid = threadIdx.x;
    if (tid < N_GT) { rowS[tid] = 0ull; posS[tid] = 0ull; negS[tid] = 0ull; }

    if (tid < 32) {  // wave 0 issues the TDM load (EXEC ignored by tensor ops)
        unsigned lds_addr = (unsigned)(unsigned long long)(void*)&gtS[0];
        unsigned long long ga = (unsigned long long)(const void*)gt;
        u32x4 g0;
        g0[0] = 1u;                                    // count=1, user descriptor
        g0[1] = lds_addr;                              // LDS destination (bytes)
        g0[2] = (unsigned)ga;                          // global_addr[31:0]
        g0[3] = (unsigned)(ga >> 32) | (2u << 30);     // global_addr[56:32], type=2
        i32x8 g1;
        g1[0] = (int)(2u << 16);                       // data_size=4B, no multicast
        g1[1] = (int)(4u << 16);                       // tensor_dim0 = 4 elements
        g1[2] = (int)(128u << 16);                     // tensor_dim1 = 128 rows
        g1[3] = (int)(4u << 16);                       // tile_dim0 = 4
        g1[4] = 128;                                   // tile_dim1 = 128
        g1[5] = 4;                                     // tensor_dim0_stride = 4
        g1[6] = 0;
        g1[7] = 0;
        i32x4 z4; z4[0] = z4[1] = z4[2] = z4[3] = 0;   // groups 2/3 unused (2D)
        i32x8 z8;
        #pragma unroll
        for (int q = 0; q < 8; ++q) z8[q] = 0;
        __builtin_amdgcn_tensor_load_to_lds(g0, g1, z4, z4, z8, 0);
        __builtin_amdgcn_s_wait_tensorcnt(0);
    }
    __syncthreads();

    const int j = blockIdx.x * blockDim.x + tid;
    if (j < M_PROPS) {
        float4 p = proposals[j];
        float ap = (p.z - p.x) * (p.w - p.y);
        unsigned long long bpos = 0ull, bneg = 0ull;
        #pragma unroll 4
        for (int r = 0; r < N_GT; ++r) {
            float4 g = gtS[r];
            float ag = (g.z - g.x) * (g.w - g.y);
            float iw = fminf(g.z, p.z) - fmaxf(g.x, p.x);
            float ih = fminf(g.w, p.w) - fmaxf(g.y, p.y);
            iw = fmaxf(iw, 0.0f); ih = fmaxf(ih, 0.0f);
            float inter = iw * ih;
            float iou = inter / (ag + ap - inter);
            unsigned flat = (unsigned)r * (unsigned)M_PROPS + (unsigned)j;

            unsigned long long rkey = packKey(__float_as_uint(iou), (unsigned)j);
            if (rkey > rowS[r]) atomicMax(&rowS[r], rkey);   // ds_max_u64

            if (iou > POS_THR) {
                unsigned long long k = packKey(hash32(flat, SALT_POS), flat);
                if (k > bpos) bpos = k;
            }
            if (iou < NEG_THR) {
                unsigned long long k = packKey(hash32(flat, SALT_NEG), flat);
                if (k > bneg) bneg = k;
            }
        }
        if (bpos) atomicMax(&posS[hash32((unsigned)bpos, SALT_BKT) & (N_GT - 1)], bpos);
        if (bneg) atomicMax(&negS[hash32((unsigned)bneg, SALT_BKT) & (N_GT - 1)], bneg);
    }
    __syncthreads();

    if (tid < N_GT) {  // one global atomic per slot per workgroup
        unsigned long long v;
        if ((v = rowS[tid]) != 0ull) atomicMax(&ws[tid],            v);
        if ((v = posS[tid]) != 0ull) atomicMax(&ws[N_GT + tid],     v);
        if ((v = negS[tid]) != 0ull) atomicMax(&ws[2 * N_GT + tid], v);
    }
}

// ---------------------------------------------------------------------------
// Kernel 3: single wave32. Seed positive buckets with per-row best pairs
// (guarantees num_pos filled, as in the reference), compute BCE + SmoothL1
// terms, reduce the 4 sums on the matrix unit, emit the scalar loss.
// ---------------------------------------------------------------------------
__global__ void __launch_bounds__(32)
finalize(const float* __restrict__ obj,
         const float4* __restrict__ proposals,
         const float4* __restrict__ gt,
         const unsigned long long* __restrict__ ws,
         float* __restrict__ out) {
    const int lane = threadIdx.x;  // 0..31, one wave

    v16h a_bce, a_w, a_sl, a_wp, ones;
    #pragma unroll
    for (int i = 0; i < 16; ++i) {
        a_bce[i] = (_Float16)0.0f; a_w[i]  = (_Float16)0.0f;
        a_sl[i]  = (_Float16)0.0f; a_wp[i] = (_Float16)0.0f;
        ones[i]  = (_Float16)1.0f;
    }

    #pragma unroll
    for (int i = 0; i < 4; ++i) {
        const int k = lane * 4 + i;  // sample slot 0..127

        // ---- positive: max(bucket winner, seeded per-row-best candidate) ----
        unsigned long long rb = ws[k];
        unsigned best_flat = (unsigned)k * (unsigned)M_PROPS + (unsigned)(rb & 0xFFFFFFFFu);
        unsigned long long seed = packKey(hash32((unsigned)k, SALT_BEST), best_flat);
        unsigned long long pk = ws[N_GT + k];
        if (seed > pk) pk = seed;

        unsigned pflat = (unsigned)pk;
        unsigned prow  = pflat / (unsigned)M_PROPS;
        unsigned pcol  = pflat - prow * (unsigned)M_PROPS;

        float x = obj[pcol];
        float bce_p = fmaxf(x, 0.0f) - x + log1pf(__expf(-fabsf(x)));  // label=1
        a_bce[i] = (_Float16)bce_p;
        a_w[i]   = (_Float16)1.0f;
        a_wp[i]  = (_Float16)1.0f;

        float4 p = proposals[pcol];
        float4 g = gt[prow];
        a_sl[4 * i + 0] = (_Float16)smooth_l1(p.x - g.x);
        a_sl[4 * i + 1] = (_Float16)smooth_l1(p.y - g.y);
        a_sl[4 * i + 2] = (_Float16)smooth_l1(p.z - g.z);
        a_sl[4 * i + 3] = (_Float16)smooth_l1(p.w - g.w);

        // ---- negative ----
        unsigned long long nk = ws[2 * N_GT + k];
        float wn = (nk != 0ull) ? 1.0f : 0.0f;
        unsigned ncol = ((unsigned)nk) % (unsigned)M_PROPS;
        float xn = obj[ncol];
        float bce_n = fmaxf(xn, 0.0f) + log1pf(__expf(-fabsf(xn)));    // label=0
        a_bce[8 + i] = (_Float16)(bce_n * wn);
        a_w[8 + i]   = (_Float16)wn;
    }

    // uniform control flow: all 32 lanes execute the WMMAs (EXEC all-ones)
    float s_bce = wmma_sum512(a_bce, ones);
    float s_w   = wmma_sum512(a_w,   ones);
    float s_sl  = wmma_sum512(a_sl,  ones);
    float s_wp  = wmma_sum512(a_wp,  ones);

    if (lane == 0) {
        float cls = s_bce / fmaxf(s_w, 1.0f);
        float reg = s_sl  / fmaxf(s_wp * 4.0f, 1.0f);
        out[0] = cls + REG_FACT * reg;
    }
}

extern "C" void kernel_launch(void* const* d_in, const int* in_sizes, int n_in,
                              void* d_out, int out_size, void* d_ws, size_t ws_size,
                              hipStream_t stream) {
    (void)in_sizes; (void)n_in; (void)out_size; (void)ws_size;
    const float*  obj       = (const float*)d_in[0];
    const float4* proposals = (const float4*)d_in[1];
    const float4* gtb       = (const float4*)d_in[2];
    float* out = (float*)d_out;
    unsigned long long* ws = (unsigned long long*)d_ws;

    init_ws<<<1, 3 * N_GT, 0, stream>>>(ws);

    const int threads = 256;
    const int blocks = (M_PROPS + threads - 1) / threads;
    iou_sample<<<blocks, threads, 0, stream>>>(proposals, gtb, ws);

    finalize<<<1, 32, 0, stream>>>(obj, proposals, gtb, ws, out);
}
